// ClusterGCN_22033182228600
// MI455X (gfx1250) — matrix operand order repaired and checked by reading, not compiled, not run on hardware
//
#include <hip/hip_runtime.h>

// ---------------------------------------------------------------------------
// GraphSAGE (3 layers) for MI455X / gfx1250, wave32 + WMMA bf16.
//   h = relu( (mean-agg(h) @ Wl + b) + h @ Wr )   x2, then final layer no relu
// GEMMs run as bf16 WMMA (f32 accum) over the concatenated K=1024 operand.
// Aggregation = f32 global atomics (agg buffer fits in 192MB L2); the scaled
// aggregation is converted to bf16 ONCE per layer so the WMMA K-loop is pure
// b128 loads + v_wmma (no per-tile cvt VALU work).
// ---------------------------------------------------------------------------

typedef __bf16        v16bf __attribute__((ext_vector_type(16)));
typedef float         v8f   __attribute__((ext_vector_type(8)));
typedef unsigned int  u32x4v __attribute__((ext_vector_type(4)));
typedef unsigned int  u32x8v __attribute__((ext_vector_type(8)));

#define DF   512      // feature width of hidden layers / input
#define NPAD 50176    // 50000 padded to multiple of 512 (and of 128)

// ---------------- small helpers ----------------
__device__ __forceinline__ unsigned short f2bf(float f) {
  union { float f; unsigned u; } c; c.f = f;
  unsigned u = c.u;
  unsigned r = (u + 0x7FFFu + ((u >> 16) & 1u)) >> 16;   // round-nearest-even
  return (unsigned short)r;
}
__device__ __forceinline__ float bf2f(unsigned short s) {
  union { unsigned u; float f; } c; c.u = ((unsigned)s) << 16; return c.f;
}
__device__ __forceinline__ unsigned pk2(float a, float b) {
  return (unsigned)f2bf(a) | ((unsigned)f2bf(b) << 16);
}
__device__ __forceinline__ void atomAddF(float* p, float v) {
  __hip_atomic_fetch_add(p, v, __ATOMIC_RELAXED, __HIP_MEMORY_SCOPE_AGENT);
}

// ---------------- utility kernels ----------------
__global__ void zero_f32(float4* p, long long n4) {
  for (long long i = blockIdx.x * (long long)blockDim.x + threadIdx.x; i < n4;
       i += (long long)gridDim.x * blockDim.x)
    p[i] = make_float4(0.f, 0.f, 0.f, 0.f);
}

__global__ void deg_count(const long long* __restrict__ dst, float* __restrict__ deg, int E) {
  int e = blockIdx.x * blockDim.x + threadIdx.x;
  if (e < E) atomAddF(&deg[(int)dst[e]], 1.0f);
}

__global__ void deg_inv_k(float* deg, int n) {
  int i = blockIdx.x * blockDim.x + threadIdx.x;
  if (i < n) deg[i] = 1.0f / fmaxf(deg[i], 1.0f);
}

// x (f32, N x 512) -> bf16 padded (NPAD x 512); pad rows zeroed
__global__ void cvt_x(const float* __restrict__ x, unsigned short* __restrict__ out, int N) {
  const long long total = (long long)NPAD * (DF / 2);
  for (long long i = blockIdx.x * (long long)blockDim.x + threadIdx.x; i < total;
       i += (long long)gridDim.x * blockDim.x) {
    int row = (int)(i >> 8);           // DF/2 = 256 uints per row
    int f   = ((int)i & 255) << 1;
    float a = 0.f, b = 0.f;
    if (row < N) {
      float2 v = *(const float2*)(x + (long long)row * DF + f);
      a = v.x; b = v.y;
    }
    ((unsigned*)out)[i] = pk2(a, b);
  }
}

// one block (256 thr) per edge: each thread scatters 2 features via f32 atomics
__global__ void agg_edges(const unsigned short* __restrict__ feat,
                          const long long* __restrict__ src,
                          const long long* __restrict__ dst,
                          float* __restrict__ agg, int E) {
  int e = blockIdx.x;
  if (e >= E) return;
  int s = (int)src[e], d = (int)dst[e];
  unsigned u = ((const unsigned*)(feat + (long long)s * DF))[threadIdx.x];
  float* ap = agg + (long long)d * DF + threadIdx.x * 2;
  atomAddF(ap,     bf2f((unsigned short)(u & 0xFFFFu)));
  atomAddF(ap + 1, bf2f((unsigned short)(u >> 16)));
}

// scale by deg_inv AND convert to bf16 in one pass (hoists all cvt math out
// of the WMMA K-loop; also halves GEMM A-operand traffic)
__global__ void scale_cvt(const float4* __restrict__ agg,
                          const float* __restrict__ deginv,
                          unsigned short* __restrict__ aggB, long long n4) {
  for (long long i = blockIdx.x * (long long)blockDim.x + threadIdx.x; i < n4;
       i += (long long)gridDim.x * blockDim.x) {
    float s = deginv[(int)(i >> 7)];   // DF/4 = 128 float4 per row
    float4 v = agg[i];
    uint2 o;
    o.x = pk2(v.x * s, v.y * s);
    o.y = pk2(v.z * s, v.w * s);
    ((uint2*)aggB)[i] = o;
  }
}

// Pack [Wl;Wr] (K=2*Din rows x Dout cols, f32) into the wave32 bf16 B-fragment
// layout: element id -> (t = K-tile of 32, nt = N-tile of 16, lane, i) with
// value = W[k = t*32 + 8*(lane>=16) + (i&7) + 16*(i>=8)][col = nt*16 + (lane&15)].
__global__ void pack_w(const float* __restrict__ Wl, const float* __restrict__ Wr,
                       unsigned short* __restrict__ Bp, int Din, int Dout) {
  int total = 2 * Din * Dout;
  int NT = Dout >> 4;
  for (int id = blockIdx.x * blockDim.x + threadIdx.x; id < total;
       id += gridDim.x * blockDim.x) {
    int i    = id & 15;
    int lane = (id >> 4) & 31;
    int tn   = id >> 9;
    int nt   = tn % NT;
    int t    = tn / NT;
    int k    = t * 32 + ((lane >> 4) << 3) + (i & 7) + ((i >> 3) << 4);
    int col  = nt * 16 + (lane & 15);
    float v  = (k < Din) ? Wl[(long long)k * Dout + col]
                         : Wr[(long long)(k - Din) * Dout + col];
    Bp[id] = f2bf(v);
  }
}

// ---------------- fragment loaders ----------------
// A (16x32 bf16) layout: lane holds row (lane&15), K chunks [kofs,kofs+8) and
// [kofs+16,kofs+24), kofs = 8*(lane>>4).  p points at row + k0 + kofs.
__device__ __forceinline__ v16bf ldA_bf16(const unsigned short* p) {
  u32x4v c0 = *(const u32x4v*)p;
  u32x4v c1 = *(const u32x4v*)(p + 16);
  u32x8v u;
  u[0] = c0[0]; u[1] = c0[1]; u[2] = c0[2]; u[3] = c0[3];
  u[4] = c1[0]; u[5] = c1[1]; u[6] = c1[2]; u[7] = c1[3];
  return __builtin_bit_cast(v16bf, u);
}
__device__ __forceinline__ v16bf ldB(const unsigned short* p) {
  u32x8v u = *(const u32x8v*)p;
  return __builtin_bit_cast(v16bf, u);
}

// ---------------- fused SAGE GEMM ----------------
// out[N, Dout] = act( [aggB | root] @ Bpack + bias )
// block = 256 thr = 8 waves (4 in M x 2 in N); block tile 128x64;
// each wave: 32x32 tile = 2x2 WMMA accumulators; K loop 1024/32 = 32 steps.
__global__ __launch_bounds__(256) void sage_gemm(
    const unsigned short* __restrict__ AaggB,  // [NPAD, 512] bf16 (pre-scaled)
    const unsigned short* __restrict__ Aroot,  // [NPAD, 512] bf16
    const unsigned short* __restrict__ Bpack,  // packed weights, K=1024
    const float* __restrict__ bias,            // [Dout]
    float* __restrict__ outF,                  // f32 out (final layer) or null
    unsigned short* __restrict__ outB,         // bf16 out (hidden) or null
    int Dout, int Nvalid, int doRelu) {
  const int lane = threadIdx.x & 31;
  const int w    = threadIdx.x >> 5;
  const int wm   = w >> 1;
  const int wn   = w & 1;
  const int rowBase = blockIdx.x * 128 + wm * 32;
  const int colBase = blockIdx.y * 64 + wn * 32;
  const int r    = lane & 15;
  const int half = lane >> 4;
  const int kofs = half * 8;
  const int NT   = Dout >> 4;
  const long long bstride = (long long)NT * 512;   // ushorts per K-tile

  v8f acc00 = {0.f,0.f,0.f,0.f,0.f,0.f,0.f,0.f};
  v8f acc01 = {0.f,0.f,0.f,0.f,0.f,0.f,0.f,0.f};
  v8f acc10 = {0.f,0.f,0.f,0.f,0.f,0.f,0.f,0.f};
  v8f acc11 = {0.f,0.f,0.f,0.f,0.f,0.f,0.f,0.f};

  const unsigned short* bp =
      Bpack + ((long long)(colBase >> 4) * 32 + lane) * 16;

  // ---- K = 0..511 : scaled neighbor aggregation (bf16) ----
  {
    const unsigned short* a0p = AaggB + (long long)(rowBase + r) * DF + kofs;
    const unsigned short* a1p = AaggB + (long long)(rowBase + 16 + r) * DF + kofs;
#pragma unroll 4
    for (int t = 0; t < DF / 32; ++t) {
      if (t + 1 < DF / 32) {
        __builtin_prefetch(a0p + (t + 1) * 32, 0, 1);
        __builtin_prefetch(a1p + (t + 1) * 32, 0, 1);
      }
      v16bf a0 = ldA_bf16(a0p + t * 32);
      v16bf a1 = ldA_bf16(a1p + t * 32);
      const unsigned short* bptr = bp + (long long)t * bstride;
      v16bf b0 = ldB(bptr);
      v16bf b1 = ldB(bptr + 512);
      acc00 = __builtin_amdgcn_wmma_f32_16x16x32_bf16(false, a0, false, b0, (short)0, acc00, false, false);
      acc01 = __builtin_amdgcn_wmma_f32_16x16x32_bf16(false, a0, false, b1, (short)0, acc01, false, false);
      acc10 = __builtin_amdgcn_wmma_f32_16x16x32_bf16(false, a1, false, b0, (short)0, acc10, false, false);
      acc11 = __builtin_amdgcn_wmma_f32_16x16x32_bf16(false, a1, false, b1, (short)0, acc11, false, false);
    }
  }
  // ---- K = 512..1023 : root features (bf16) ----
  {
    const unsigned short* r0p = Aroot + (long long)(rowBase + r) * DF + kofs;
    const unsigned short* r1p = Aroot + (long long)(rowBase + 16 + r) * DF + kofs;
#pragma unroll 4
    for (int t = 0; t < DF / 32; ++t) {
      if (t + 1 < DF / 32) {
        __builtin_prefetch(r0p + (t + 1) * 32, 0, 1);
        __builtin_prefetch(r1p + (t + 1) * 32, 0, 1);
      }
      v16bf a0 = ldA_bf16(r0p + t * 32);
      v16bf a1 = ldA_bf16(r1p + t * 32);
      const unsigned short* bptr = bp + (long long)(t + DF / 32) * bstride;
      v16bf b0 = ldB(bptr);
      v16bf b1 = ldB(bptr + 512);
      acc00 = __builtin_amdgcn_wmma_f32_16x16x32_bf16(false, a0, false, b0, (short)0, acc00, false, false);
      acc01 = __builtin_amdgcn_wmma_f32_16x16x32_bf16(false, a0, false, b1, (short)0, acc01, false, false);
      acc10 = __builtin_amdgcn_wmma_f32_16x16x32_bf16(false, a1, false, b0, (short)0, acc10, false, false);
      acc11 = __builtin_amdgcn_wmma_f32_16x16x32_bf16(false, a1, false, b1, (short)0, acc11, false, false);
    }
  }

  // ---- epilogue: bias (+relu), C/D layout: row = 8*half + v, col = lane&15 ----
  float bias0 = bias[colBase + r];
  float bias1 = bias[colBase + 16 + r];
#pragma unroll
  for (int v = 0; v < 8; ++v) {
    int m0 = rowBase + half * 8 + v;
    int m1 = m0 + 16;
    int c0 = colBase + r;
    int c1 = c0 + 16;
    float o00 = acc00[v] + bias0;
    float o01 = acc01[v] + bias1;
    float o10 = acc10[v] + bias0;
    float o11 = acc11[v] + bias1;
    if (doRelu) {
      o00 = fmaxf(o00, 0.f); o01 = fmaxf(o01, 0.f);
      o10 = fmaxf(o10, 0.f); o11 = fmaxf(o11, 0.f);
    }
    if (outB) {
      outB[(long long)m0 * Dout + c0] = f2bf(o00);
      outB[(long long)m0 * Dout + c1] = f2bf(o01);
      outB[(long long)m1 * Dout + c0] = f2bf(o10);
      outB[(long long)m1 * Dout + c1] = f2bf(o11);
    } else {
      if (m0 < Nvalid) {
        outF[(long long)m0 * Dout + c0] = o00;
        outF[(long long)m0 * Dout + c1] = o01;
      }
      if (m1 < Nvalid) {
        outF[(long long)m1 * Dout + c0] = o10;
        outF[(long long)m1 * Dout + c1] = o11;
      }
    }
  }
}

// ---------------- host orchestration ----------------
extern "C" void kernel_launch(void* const* d_in, const int* in_sizes, int n_in,
                              void* d_out, int out_size, void* d_ws, size_t ws_size,
                              hipStream_t stream) {
  const float*     x   = (const float*)d_in[0];
  const long long* ei  = (const long long*)d_in[1];
  // d_in[2] = relations (unused, as in reference)
  const float* Wl0 = (const float*)d_in[3];
  const float* Wr0 = (const float*)d_in[4];
  const float* b0  = (const float*)d_in[5];
  const float* Wl1 = (const float*)d_in[6];
  const float* Wr1 = (const float*)d_in[7];
  const float* b1  = (const float*)d_in[8];
  const float* Wl2 = (const float*)d_in[9];
  const float* Wr2 = (const float*)d_in[10];
  const float* b2  = (const float*)d_in[11];

  const int E = in_sizes[1] / 2;
  const int N = in_sizes[0] / DF;

  // workspace layout (~258 MB)
  char* ws = (char*)d_ws;
  size_t o = 0;
  float* deg = (float*)(ws + o);                  o += (size_t)NPAD * 4;
  float* agg = (float*)(ws + o);                  o += (size_t)NPAD * DF * 4;
  unsigned short* aggB = (unsigned short*)(ws + o); o += (size_t)NPAD * DF * 2;
  unsigned short* hA = (unsigned short*)(ws + o);   o += (size_t)NPAD * DF * 2;
  unsigned short* hB = (unsigned short*)(ws + o);   o += (size_t)NPAD * DF * 2;
  unsigned short* Bp = (unsigned short*)(ws + o);   // 2*512*512*2 = 1 MB max

  const long long* srcIdx = ei;
  const long long* dstIdx = ei + E;

  const long long aggN4 = (long long)NPAD * DF / 4;
  const int gx = (N + 127) / 128;

  // degrees + deg_inv (computed once, reused for all 3 layers)
  zero_f32<<<512, 256, 0, stream>>>((float4*)deg, NPAD / 4);
  deg_count<<<(E + 255) / 256, 256, 0, stream>>>(dstIdx, deg, E);
  deg_inv_k<<<(NPAD + 255) / 256, 256, 0, stream>>>(deg, NPAD);

  // x -> bf16 (hA)
  cvt_x<<<2048, 256, 0, stream>>>(x, hA, N);

  // ---- layer 0: in hA -> out hB ----
  zero_f32<<<4096, 256, 0, stream>>>((float4*)agg, aggN4);
  agg_edges<<<E, 256, 0, stream>>>(hA, srcIdx, dstIdx, agg, E);
  scale_cvt<<<4096, 256, 0, stream>>>((const float4*)agg, deg, aggB, aggN4);
  pack_w<<<1024, 256, 0, stream>>>(Wl0, Wr0, Bp, DF, DF);
  sage_gemm<<<dim3(gx, DF / 64), 256, 0, stream>>>(aggB, hA, Bp, b0,
                                                   nullptr, hB, DF, N, 1);

  // ---- layer 1: in hB -> out hA ----
  zero_f32<<<4096, 256, 0, stream>>>((float4*)agg, aggN4);
  agg_edges<<<E, 256, 0, stream>>>(hB, srcIdx, dstIdx, agg, E);
  scale_cvt<<<4096, 256, 0, stream>>>((const float4*)agg, deg, aggB, aggN4);
  pack_w<<<1024, 256, 0, stream>>>(Wl1, Wr1, Bp, DF, DF);
  sage_gemm<<<dim3(gx, DF / 64), 256, 0, stream>>>(aggB, hB, Bp, b1,
                                                   nullptr, hA, DF, N, 1);

  // ---- layer 2: in hA -> out d_out (f32, Dout=64, no relu) ----
  zero_f32<<<4096, 256, 0, stream>>>((float4*)agg, aggN4);
  agg_edges<<<E, 256, 0, stream>>>(hA, srcIdx, dstIdx, agg, E);
  scale_cvt<<<4096, 256, 0, stream>>>((const float4*)agg, deg, aggB, aggN4);
  pack_w<<<256, 256, 0, stream>>>(Wl2, Wr2, Bp, DF, 64);
  sage_gemm<<<dim3(gx, 1), 256, 0, stream>>>(aggB, hA, Bp, b2,
                                             (float*)d_out, nullptr, 64, N, 0);
}
